// InterpretableMultiHeadAttention_68599217652303
// MI455X (gfx1250) — matrix operand order repaired and checked by reading, not compiled, not run on hardware
//
#include <hip/hip_runtime.h>
#include <hip/hip_bf16.h>
#include <cstdint>

// InterpretableMultiHeadAttention for MI455X (gfx1250, wave32, WMMA).
// B=2, S=2048, D=1024, H=16, DK=64.
#define B_  2
#define S_  2048
#define D_  1024
#define H_  16
#define DK_ 64

typedef __attribute__((ext_vector_type(16))) _Float16 v16h;
typedef __attribute__((ext_vector_type(8)))  _Float16 v8h;
typedef __attribute__((ext_vector_type(8)))  float    v8f;

static __device__ __forceinline__ v8f wmma_f16(v16h a, v16h b, v8f c) {
  // D = A(16x32 f16) * B(32x16 f16) + C(16x16 f32)
  return __builtin_amdgcn_wmma_f32_16x16x32_f16(false, a, false, b, (short)0, c,
                                                false, false);
}
static __device__ __forceinline__ v16h cat8(v8h lo, v8h hi) {
  return __builtin_shufflevector(lo, hi, 0, 1, 2, 3, 4, 5, 6, 7,
                                 8, 9, 10, 11, 12, 13, 14, 15);
}

// ---------------------------------------------------------------------------
// Kernel P: one-time transpose+convert of weights to f16 B-frag layout.
//   z 0..15 : Wq[h] (D x DK) -> Wt[z]  (DK x D)
//   z 16..31: Wk[h] (D x DK) -> Wt[z]  (DK x D)
//   z 32    : Wv    (D x DK) -> Wt[32] (DK x D)
//   z 33    : Wh    (DK x D) -> Wht    (D x DK)
// ---------------------------------------------------------------------------
__global__ __launch_bounds__(256) void transpose_conv_kernel(
    const float* __restrict__ Wq, const float* __restrict__ Wk,
    const float* __restrict__ Wv, const float* __restrict__ Wh,
    _Float16* __restrict__ Wt, _Float16* __restrict__ Wht) {
  __shared__ float tile[32][33];
  const int z = blockIdx.z;
  const float* src; _Float16* dst; int R, C;
  if (z < 16)      { src = Wq + (size_t)z * D_ * DK_;
                     dst = Wt + (size_t)z * D_ * DK_;        R = D_;  C = DK_; }
  else if (z < 32) { src = Wk + (size_t)(z - 16) * D_ * DK_;
                     dst = Wt + (size_t)z * D_ * DK_;        R = D_;  C = DK_; }
  else if (z == 32){ src = Wv;
                     dst = Wt + (size_t)32 * D_ * DK_;       R = D_;  C = DK_; }
  else             { src = Wh; dst = Wht;                    R = DK_; C = D_;  }
  const int c0 = blockIdx.x * 32, r0 = blockIdx.y * 32;
  if (c0 >= C || r0 >= R) return;
  const int tx = threadIdx.x, ty = threadIdx.y;  // block (32, 8)
  #pragma unroll
  for (int i = 0; i < 4; ++i)
    tile[ty + 8 * i][tx] = src[(size_t)(r0 + ty + 8 * i) * C + c0 + tx];
  __syncthreads();
  #pragma unroll
  for (int i = 0; i < 4; ++i)
    dst[(size_t)(c0 + ty + 8 * i) * R + r0 + tx] = (_Float16)tile[tx][ty + 8 * i];
}

// ---------------------------------------------------------------------------
// Kernel A: fused projection GEMMs, LDS-free (operands are L2-resident).
//   job 0..15  : qs[h]  = q @ Wq[h] + bq[h]   (4096x64, K=1024)
//   job 16..31 : ksh[h] = k @ Wk[h] + bk[h]
//   job 32     : vst    = (v @ Wv + bv)^T     (stored [b][n][t])
// Block = 128 threads (4 wave32s); each wave owns a 16(M) x 64(N) tile.
// Software-pipelined: next k-step activations load under current WMMAs;
// all 4 B-fragments issued as one clause ahead of the 4 WMMAs.
// ---------------------------------------------------------------------------
__global__ __launch_bounds__(128) void proj_gemm_kernel(
    const float* __restrict__ q, const float* __restrict__ k,
    const float* __restrict__ v,
    const _Float16* __restrict__ Wt,
    const float* __restrict__ bq, const float* __restrict__ bk,
    const float* __restrict__ bv,
    _Float16* __restrict__ qs, _Float16* __restrict__ ksh,
    _Float16* __restrict__ vst) {
  const int tid  = threadIdx.x;
  const int lane = tid & 31;
  const int w    = tid >> 5;
  const int job  = blockIdx.y;
  const int gr0  = blockIdx.x * 64;

  const float* X; const float* bias; int h = 0; bool flat = false;
  if (job < H_)        { X = q; bias = bq + job * DK_;        h = job; }
  else if (job < 2*H_) { X = k; bias = bk + (job - H_) * DK_; h = job - H_; }
  else                 { X = v; bias = bv; flat = true; }
  const _Float16* Wjob = Wt + (size_t)job * D_ * DK_;  // (DK x D) f16

  const int ml   = lane & 15;
  const int hh   = lane >> 4;
  const int kb8  = hh * 8;
  const int kb16 = hh * 16;
  const float* xrow = X + (size_t)(gr0 + w * 16 + ml) * D_;
  const _Float16* wrow = Wjob + (size_t)ml * D_ + kb16;

  v8f zero = {};
  v8f c[4] = {zero, zero, zero, zero};

  v8f xa = *reinterpret_cast<const v8f*>(xrow + kb16);
  v8f xb = *reinterpret_cast<const v8f*>(xrow + kb16 + 8);

  for (int k0 = 0; k0 < D_; k0 += 32) {
    // B-fragments for this k-step: 4 x 32B contiguous loads (one clause)
    v16h bf[4];
    #pragma unroll
    for (int nt = 0; nt < 4; ++nt)
      bf[nt] = *reinterpret_cast<const v16h*>(wrow + (size_t)(nt * 16) * D_ + k0);

    // A-fragment from activations prefetched last iteration
    v16h a;
    #pragma unroll
    for (int i = 0; i < 8; ++i) { a[i] = (_Float16)xa[i]; a[8 + i] = (_Float16)xb[i]; }

    // prefetch next k-step activations under the WMMAs
    if (k0 + 32 < D_) {
      xa = *reinterpret_cast<const v8f*>(xrow + k0 + 32 + kb16);
      xb = *reinterpret_cast<const v8f*>(xrow + k0 + 40 + kb16);
    }

    #pragma unroll
    for (int nt = 0; nt < 4; ++nt) c[nt] = wmma_f16(a, bf[nt], c[nt]);
  }

  const int mh = hh * 8;
  const int sbase = gr0 + w * 16 + mh;   // 8 consecutive output rows
  #pragma unroll
  for (int nt = 0; nt < 4; ++nt) {
    const int n = nt * 16 + ml;
    const float bval = bias[n];
    if (flat) {
      // vst[b][n][t]: per-lane the 8 rows are contiguous t -> one v8h store
      const int bb = sbase >> 11, ss = sbase & (S_ - 1);
      v8h hv;
      #pragma unroll
      for (int r = 0; r < 8; ++r) hv[r] = (_Float16)(c[nt][r] + bval);
      *reinterpret_cast<v8h*>(vst + ((size_t)bb * DK_ + n) * S_ + ss) = hv;
    } else {
      _Float16* outp = (job < H_) ? qs : ksh;
      #pragma unroll
      for (int r = 0; r < 8; ++r) {
        const int gr = sbase + r;
        const int bb = gr >> 11, ss = gr & (S_ - 1);
        outp[(((size_t)bb * H_ + h) * S_ + ss) * DK_ + n] =
            (_Float16)(c[nt][r] + bval);
      }
    }
  }
}

// ---------------------------------------------------------------------------
// Kernel B: fused scores + sigmoid-blend + softmax + attn write + attn@V.
// One block (8 wave32s) per (b, h, 16-row s-tile). Blended-score stripe
// (16 x 2048 f16 = 64KB) lives in CDNA5's 320KB LDS.
// ---------------------------------------------------------------------------
__global__ __launch_bounds__(256) void attn_kernel(
    const _Float16* __restrict__ qs, const _Float16* __restrict__ ksh,
    const _Float16* __restrict__ vst,
    const unsigned char* __restrict__ mask, const float* __restrict__ prior,
    float* __restrict__ attn_out, float* __restrict__ heads_mean) {
  __shared__ _Float16 bl[16][S_];       // 64 KB blended-score stripe
  __shared__ float    heads_s[16][DK_]; // 4 KB cross-wave head accumulator
  __shared__ float    wM[8][16], wL[8][16];
  __shared__ float    fM[16], fLinv[16];

  const int tid  = threadIdx.x;
  const int lane = tid & 31;
  const int w    = tid >> 5;
  const int nst  = S_ / 16;                       // 128 s-tiles
  const int st   = blockIdx.x % nst;
  const int bh   = blockIdx.x / nst;
  const int b    = bh / H_;
  const int h    = bh % H_;
  const int s0   = st * 16;

  for (int idx = tid; idx < 16 * DK_; idx += 256) heads_s[idx >> 6][idx & 63] = 0.0f;

  const int ml   = lane & 15;
  const int hh   = lane >> 4;
  const int kb8  = hh * 8;
  const int kb16 = hh * 16;
  const int mh   = hh * 8;

  // qs A-frags straight from global (L2-resident, 2KB per tile)
  const _Float16* qrow = qs + ((size_t)bh * S_ + s0 + ml) * DK_;
  const v16h a0 = cat8(*reinterpret_cast<const v8h*>(qrow + kb16),
                       *reinterpret_cast<const v8h*>(qrow + kb16 + 8));
  const v16h a1 = cat8(*reinterpret_cast<const v8h*>(qrow + 32 + kb16),
                       *reinterpret_cast<const v8h*>(qrow + 40 + kb16));

  float mx[8], sm[8];
  #pragma unroll
  for (int r = 0; r < 8; ++r) { mx[r] = -1.0e30f; sm[r] = 0.0f; }

  const size_t mpbase = (size_t)b * S_ * S_ + (size_t)s0 * S_ + ml;
  const _Float16* kbase = ksh + ((size_t)bh * S_ + ml) * DK_ + kb16;

  // ---- Phase 1: scores -> blend -> LDS, online row max/sum -------------
  // ksh fragment loads are pipelined one t-tile ahead so they complete
  // under the exp-heavy elementwise section of the previous tile.
  v16h bk0 = *reinterpret_cast<const v16h*>(kbase + (size_t)(w * 16) * DK_);
  v16h bk1 = *reinterpret_cast<const v16h*>(kbase + (size_t)(w * 16) * DK_ + 32);
  for (int tt = w; tt < S_ / 16; tt += 8) {
    const int t0 = tt * 16;
    v8f c = {};
    c = wmma_f16(a0, bk0, c);
    c = wmma_f16(a1, bk1, c);
    if (tt + 8 < S_ / 16) {
      bk0 = *reinterpret_cast<const v16h*>(kbase + (size_t)(t0 + 128) * DK_);
      bk1 = *reinterpret_cast<const v16h*>(kbase + (size_t)(t0 + 128) * DK_ + 32);
    }
    // mask/prior for this tile issued before the elementwise chain
    float pr[8]; unsigned char mk[8];
    #pragma unroll
    for (int r = 0; r < 8; ++r) {
      const size_t mi = mpbase + (size_t)(mh + r) * S_ + t0;
      pr[r] = prior[mi];
      mk[r] = mask[mi];
    }
    #pragma unroll
    for (int r = 0; r < 8; ++r) {
      float sc = c[r] * 0.125f;                 // 1/sqrt(DK)
      float p  = pr[r];
      if (mk[r]) { sc = -1.0e9f; p = -1.0e9f; }
      float z   = 1.0f / (1.0f + __expf(-(sc + p)));
      float blv = z * sc + (1.0f - z) * p;
      bl[mh + r][t0 + ml] = (_Float16)blv;
      float nm = fmaxf(mx[r], blv);
      sm[r] = sm[r] * __expf(mx[r] - nm) + __expf(blv - nm);
      mx[r] = nm;
    }
  }
  // reduce stats over the 16 lanes that share a row (stays in half-wave)
  #pragma unroll
  for (int off = 1; off < 16; off <<= 1) {
    #pragma unroll
    for (int r = 0; r < 8; ++r) {
      float om = __shfl_xor(mx[r], off, 32);
      float os = __shfl_xor(sm[r], off, 32);
      float nm = fmaxf(mx[r], om);
      sm[r] = sm[r] * __expf(mx[r] - nm) + os * __expf(om - nm);
      mx[r] = nm;
    }
  }
  if (ml == 0) {
    #pragma unroll
    for (int r = 0; r < 8; ++r) { wM[w][mh + r] = mx[r]; wL[w][mh + r] = sm[r]; }
  }
  __syncthreads();
  if (tid < 16) {
    float M = -1.0e30f, L = 0.0f;
    for (int w2 = 0; w2 < 8; ++w2) {
      float om = wM[w2][tid], os = wL[w2][tid];
      float nm = fmaxf(M, om);
      L = L * __expf(M - nm) + os * __expf(om - nm);
      M = nm;
    }
    fM[tid]    = M;
    fLinv[tid] = 1.0f / L;
  }
  __syncthreads();

  // ---- Phase 2a: vectorized nontemporal write of normalized attn -------
  // 32B per lane, coalesced; NT keeps the 537MB stream out of L2.
  for (int idx = tid; idx < 16 * (S_ / 8); idx += 256) {
    const int m  = idx >> 8;             // 256 8-wide chunks per row
    const int t0 = (idx & 255) * 8;
    const v8h bv8 = *reinterpret_cast<const v8h*>(&bl[m][t0]);
    const float M = fM[m], Li = fLinv[m];
    v8f av;
    #pragma unroll
    for (int i = 0; i < 8; ++i) av[i] = __expf((float)bv8[i] - M) * Li;
    const size_t o = (((size_t)(b * S_ + s0 + m)) * H_ + h) * S_ + t0;
    __builtin_nontemporal_store(av, reinterpret_cast<v8f*>(attn_out + o));
  }

  // ---- Phase 2b: heads += attn @ vs (WMMA over 32-wide t chunks) -------
  v8f zero = {};
  v8f hc[4] = {zero, zero, zero, zero};
  const float Mrow = fM[ml], Lrow = fLinv[ml];
  const _Float16* vrow = vst + ((size_t)b * DK_ + ml) * S_ + kb16;
  for (int tc = w; tc < S_ / 32; tc += 8) {
    const int t0 = tc * 32;
    // vst B-frags: 4 x 32B contiguous loads issued as one clause
    v16h bf[4];
    #pragma unroll
    for (int nt = 0; nt < 4; ++nt)
      bf[nt] = *reinterpret_cast<const v16h*>(vrow + (size_t)(nt * 16) * S_ + t0);
    const v8h p0 = *reinterpret_cast<const v8h*>(&bl[ml][t0 + kb16]);
    const v8h p1 = *reinterpret_cast<const v8h*>(&bl[ml][t0 + kb16 + 8]);
    v16h af;
    #pragma unroll
    for (int i = 0; i < 8; ++i) {
      af[i]     = (_Float16)(__expf((float)p0[i] - Mrow) * Lrow);
      af[8 + i] = (_Float16)(__expf((float)p1[i] - Mrow) * Lrow);
    }
    #pragma unroll
    for (int nt = 0; nt < 4; ++nt) hc[nt] = wmma_f16(af, bf[nt], hc[nt]);
  }
  #pragma unroll
  for (int nt = 0; nt < 4; ++nt)
    #pragma unroll
    for (int r = 0; r < 8; ++r)
      atomicAdd(&heads_s[mh + r][nt * 16 + ml], hc[nt][r]);  // ds_add_f32
  __syncthreads();
  for (int idx = tid; idx < 16 * DK_; idx += 256) {
    const int m = idx >> 6, n = idx & 63;
    atomicAdd(&heads_mean[((size_t)(b * S_ + s0 + m)) * DK_ + n],
              heads_s[m][n] * (1.0f / H_));                  // global_atomic_add_f32
  }
}

// ---------------------------------------------------------------------------
// Kernel C: out = heads_mean (4096x64) @ Wh (64x1024), fp32 out. LDS-free.
// All 8 B-fragments issued before the 8 WMMAs.
// ---------------------------------------------------------------------------
__global__ __launch_bounds__(128) void out_gemm_kernel(
    const float* __restrict__ hm, const _Float16* __restrict__ Wht,
    float* __restrict__ out) {
  const int tid  = threadIdx.x;
  const int lane = tid & 31;
  const int w    = tid >> 5;
  const int gr0  = blockIdx.x * 64;
  const int n0g  = blockIdx.y * 64;

  const int ml   = lane & 15;
  const int hh   = lane >> 4;
  const int kb8  = hh * 8;
  const int kb16 = hh * 16;
  const int mh   = hh * 8;

  const float* hrow = hm + (size_t)(gr0 + w * 16 + ml) * DK_;
  v8f ha0 = *reinterpret_cast<const v8f*>(hrow + kb16);
  v8f ha1 = *reinterpret_cast<const v8f*>(hrow + kb16 + 8);
  v8f ha2 = *reinterpret_cast<const v8f*>(hrow + 32 + kb16);
  v8f ha3 = *reinterpret_cast<const v8f*>(hrow + 40 + kb16);

  const _Float16* wrow = Wht + (size_t)(n0g + ml) * DK_ + kb16;
  v16h b0[4], b1[4];
  #pragma unroll
  for (int nt = 0; nt < 4; ++nt) {
    b0[nt] = *reinterpret_cast<const v16h*>(wrow + (size_t)(nt * 16) * DK_);
    b1[nt] = *reinterpret_cast<const v16h*>(wrow + (size_t)(nt * 16) * DK_ + 32);
  }

  v16h a0, a1;
  #pragma unroll
  for (int i = 0; i < 8; ++i) {
    a0[i] = (_Float16)ha0[i]; a0[8 + i] = (_Float16)ha1[i];
    a1[i] = (_Float16)ha2[i]; a1[8 + i] = (_Float16)ha3[i];
  }

  v8f zero = {};
  v8f c[4] = {zero, zero, zero, zero};
  #pragma unroll
  for (int nt = 0; nt < 4; ++nt) {
    c[nt] = wmma_f16(a0, b0[nt], c[nt]);
    c[nt] = wmma_f16(a1, b1[nt], c[nt]);
  }
  #pragma unroll
  for (int nt = 0; nt < 4; ++nt)
    #pragma unroll
    for (int r = 0; r < 8; ++r)
      out[(size_t)(gr0 + w * 16 + mh + r) * D_ + n0g + nt * 16 + ml] = c[nt][r];
}

// ---------------------------------------------------------------------------
extern "C" void kernel_launch(void* const* d_in, const int* in_sizes, int n_in,
                              void* d_out, int out_size, void* d_ws,
                              size_t ws_size, hipStream_t stream) {
  (void)in_sizes; (void)n_in; (void)out_size; (void)ws_size;
  const float*         q     = (const float*)d_in[0];
  const float*         k     = (const float*)d_in[1];
  const float*         v     = (const float*)d_in[2];
  const unsigned char* mask  = (const unsigned char*)d_in[3];  // jax bool = 1B
  const float*         prior = (const float*)d_in[4];
  const float*         Wq    = (const float*)d_in[5];
  const float*         bq    = (const float*)d_in[6];
  const float*         Wk    = (const float*)d_in[7];
  const float*         bk    = (const float*)d_in[8];
  const float*         Wv    = (const float*)d_in[9];
  const float*         bv    = (const float*)d_in[10];
  const float*         Wh    = (const float*)d_in[11];

  // Workspace layout (~21.8 MB, fully L2-resident on MI455X's 192MB L2):
  char* ws = (char*)d_ws;
  _Float16* qs         = (_Float16*)(ws);               //  8 MB  (B,H,S,DK) f16
  _Float16* ksh        = (_Float16*)(ws + 8388608);     //  8 MB  (B,H,S,DK) f16
  _Float16* vst        = (_Float16*)(ws + 16777216);    //  0.5MB (B,DK,S)   f16
  float*    heads_mean = (float*)   (ws + 17301504);    //  1 MB  (B,S,DK)   f32
  _Float16* Wt         = (_Float16*)(ws + 18350080);    //  4.1MB 33x(DK,D)  f16
  _Float16* Wht        = (_Float16*)(ws + 22675456);    //  128KB (D,DK)     f16

  float* out_main = (float*)d_out;                       // (B,S,D) fp32
  float* attn_out = out_main + (size_t)B_ * S_ * D_;     // (B,S,H,S) fp32

  hipMemsetAsync(heads_mean, 0, (size_t)B_ * S_ * DK_ * sizeof(float), stream);

  transpose_conv_kernel<<<dim3(32, 32, 34), dim3(32, 8), 0, stream>>>(
      Wq, Wk, Wv, Wh, Wt, Wht);
  proj_gemm_kernel<<<dim3(64, 2 * H_ + 1), 128, 0, stream>>>(
      q, k, v, Wt, bq, bk, bv, qs, ksh, vst);
  attn_kernel<<<dim3(B_ * H_ * (S_ / 16)), 256, 0, stream>>>(
      qs, ksh, vst, mask, prior, attn_out, heads_mean);
  out_gemm_kernel<<<dim3(64, 16), 128, 0, stream>>>(heads_mean, Wht, out_main);
}
